// ResoneticTransformer_25374666785531
// MI455X (gfx1250) — matrix operand order repaired and checked by reading, not compiled, not run on hardware
//
#include <hip/hip_runtime.h>
#include <hip/hip_bf16.h>

// ResoneticTransformer for MI455X (gfx1250): GEMMs via v_wmma_f32_16x16x32_f16,
// tile staging via Tensor Data Mover (tensor_load_to_lds + s_wait_tensorcnt),
// flash-style attention with WMMA QK^T / PV, fp32 accumulate, f16 data movement.

#define USE_TDM 1

typedef __attribute__((ext_vector_type(16))) _Float16 v16h;
typedef __attribute__((ext_vector_type(8)))  _Float16 v8h;
typedef __attribute__((ext_vector_type(8)))  float    v8f;
typedef __attribute__((ext_vector_type(4)))  unsigned v4u;
typedef __attribute__((ext_vector_type(8)))  unsigned v8u;

union HV { v16h v16; v8h v8[2]; };

__device__ __forceinline__ v8f zero8() {
  v8f z;
  #pragma unroll
  for (int i = 0; i < 8; i++) z[i] = 0.f;
  return z;
}

__device__ __forceinline__ v8f wmma_f16(v16h a, v16h b, v8f c) {
  return __builtin_amdgcn_wmma_f32_16x16x32_f16(false, a, false, b, (short)0, c, false, false);
}

// A-fragment (16x32 f16, M x K): lane half i -> K = 8*g + i (i<8), 8*g + 16 + (i-8) (i>=8)
__device__ __forceinline__ v16h frag_a(const _Float16* base, int stride, int row, int g) {
  HV u;
  const _Float16* p = base + row * stride + 8 * g;
  u.v8[0] = *(const v8h*)p;
  u.v8[1] = *(const v8h*)(p + 16);
  return u.v16;
}

// B-fragment (32x16 f16, K x N): lane half i -> K = 8*g + i (i<8), 8*g + 16 + (i-8) (i>=8)
__device__ __forceinline__ v16h frag_b(const _Float16* base, int stride, int row, int g) {
  HV u;
  const _Float16* p = base + row * stride + 8 * g;
  u.v8[0] = *(const v8h*)p;
  u.v8[1] = *(const v8h*)(p + 16);
  return u.v16;
}

#if USE_TDM
// Issue a TDM 2D tile load (f16 elements) into LDS. Per-wave op; EXEC ignored.
// pad_code: pad_amount=4 DWORDs after every 2^(pad_interval_code+1) DWORDs, i.e.
//   tile0=32 halves (16 DW) + pad 8 halves -> LDS row stride 40 halves (ivl code 3)
//   tile0=64 halves (32 DW) + pad 8 halves -> LDS row stride 72 halves (ivl code 4)
__device__ __forceinline__ void tdm_load_2d_f16(const _Float16* gsrc, unsigned lds_byte_off,
                                                unsigned dim0_rem, unsigned dim1_rem,
                                                unsigned tile0, unsigned tile1,
                                                unsigned stride0, unsigned pad_ivl_code) {
  unsigned long long ga = (unsigned long long)(uintptr_t)gsrc;
  v4u g0;
  g0[0] = 1u;                                           // count=1, user load
  g0[1] = lds_byte_off;                                 // LDS byte address
  g0[2] = (unsigned)ga;                                 // global_addr[31:0]
  g0[3] = (unsigned)((ga >> 32) & 0x1FFFFFFu) | (2u << 30);  // addr[56:32] | type=2
  v8u g1;
  g1[0] = (1u << 16)                                    // data_size = 2 bytes
        | (1u << 20)                                    // pad_enable
        | (pad_ivl_code << 22)                          // pad_interval
        | (3u << 25);                                   // pad_amount = 4 DWORDs
  g1[1] = (dim0_rem & 0xFFFFu) << 16;                   // tensor_dim0[15:0]
  g1[2] = (dim0_rem >> 16) | ((dim1_rem & 0xFFFFu) << 16);
  g1[3] = (dim1_rem >> 16) | (tile0 << 16);             // tile_dim0
  g1[4] = tile1;                                        // tile_dim1 (tile_dim2 = 0)
  g1[5] = stride0;                                      // tensor_dim0_stride[31:0]
  g1[6] = 0u;
  g1[7] = 0u;
  asm volatile("tensor_load_to_lds %0, %1" :: "s"(g0), "s"(g1) : "memory");
}
#endif

__device__ float block_reduce_sum(float v, float* buf) {
  #pragma unroll
  for (int o = 16; o; o >>= 1) v += __shfl_down(v, o, 32);
  int lane = threadIdx.x & 31, wid = threadIdx.x >> 5;
  __syncthreads();
  if (lane == 0) buf[wid] = v;
  __syncthreads();
  if (threadIdx.x == 0) {
    float s = 0.f;
    int nw = blockDim.x >> 5;
    for (int i = 0; i < nw; i++) s += buf[i];
    buf[0] = s;
  }
  __syncthreads();
  return buf[0];
}

// ---------------------------------------------------------------- conversion
__global__ void f32_to_f16_kernel(const float* __restrict__ s, _Float16* __restrict__ d, int n) {
  for (int i = blockIdx.x * blockDim.x + threadIdx.x; i < n; i += gridDim.x * blockDim.x)
    d[i] = (_Float16)s[i];
}

// ---------------------------------------------------------------- embedding
__global__ void embed_kernel(const int* __restrict__ ids, const float* __restrict__ emb,
                             const float* __restrict__ pos, float* __restrict__ x) {
  size_t i = (size_t)blockIdx.x * blockDim.x + threadIdx.x;
  if (i >= (size_t)4096 * 512) return;
  int d = (int)(i & 511);
  int row = (int)(i >> 9);
  int l = row & 1023;
  x[i] = emb[(size_t)ids[row] * 512 + d] + pos[(size_t)l * 512 + d];
}

// --------------------------------------------------- rg gate + modulate + LN
__global__ void __launch_bounds__(256)
rg_mod_ln_kernel(const float* __restrict__ x, const float* __restrict__ rgw,
                 const float* __restrict__ rgb, const float* __restrict__ s1,
                 const float* __restrict__ b1, _Float16* __restrict__ h,
                 float* __restrict__ rg_out) {
  __shared__ float buf[8];
  const int row = blockIdx.x, t = threadIdx.x;
  const float x0 = x[(size_t)row * 512 + t], x1 = x[(size_t)row * 512 + t + 256];
  float rg[4];
  #pragma unroll
  for (int j = 0; j < 4; j++) {
    float p = x0 * rgw[j * 512 + t] + x1 * rgw[j * 512 + t + 256];
    float s = block_reduce_sum(p, buf);
    rg[j] = 1.f / (1.f + __expf(-(s + rgb[j])));
  }
  float rmean = 0.25f * (rg[0] + rg[1] + rg[2] + rg[3]);
  float xm0 = x0 * (1.f + rmean), xm1 = x1 * (1.f + rmean);
  float mean = block_reduce_sum(xm0 + xm1, buf) * (1.f / 512.f);
  float d0 = xm0 - mean, d1 = xm1 - mean;
  float var = block_reduce_sum(d0 * d0 + d1 * d1, buf) * (1.f / 512.f);
  float rs = rsqrtf(var + 1e-5f);
  h[(size_t)row * 512 + t]       = (_Float16)(d0 * rs * s1[t] + b1[t]);
  h[(size_t)row * 512 + t + 256] = (_Float16)(d1 * rs * s1[t + 256] + b1[t + 256]);
  if (rg_out && t < 4) rg_out[(size_t)row * 4 + t] = rg[t];
}

// ---------------------------------------------------------- residual + LN
__global__ void __launch_bounds__(256)
add_ln_kernel(float* __restrict__ x, const float* __restrict__ tmp,
              const float* __restrict__ s, const float* __restrict__ b,
              _Float16* __restrict__ xh) {
  __shared__ float buf[8];
  int row = blockIdx.x, t = threadIdx.x;
  float v0 = x[(size_t)row * 512 + t] + tmp[(size_t)row * 512 + t];
  float v1 = x[(size_t)row * 512 + t + 256] + tmp[(size_t)row * 512 + t + 256];
  float mean = block_reduce_sum(v0 + v1, buf) * (1.f / 512.f);
  float d0 = v0 - mean, d1 = v1 - mean;
  float var = block_reduce_sum(d0 * d0 + d1 * d1, buf) * (1.f / 512.f);
  float rs = rsqrtf(var + 1e-5f);
  float o0 = d0 * rs * s[t] + b[t], o1 = d1 * rs * s[t + 256] + b[t + 256];
  x[(size_t)row * 512 + t] = o0;
  x[(size_t)row * 512 + t + 256] = o1;
  xh[(size_t)row * 512 + t] = (_Float16)o0;
  xh[(size_t)row * 512 + t + 256] = (_Float16)o1;
}

// ---------------------------------------------------------------- phase proj
__global__ void __launch_bounds__(256)
phase_kernel(const _Float16* __restrict__ h, const float* __restrict__ pw,
             const float* __restrict__ pb, float* __restrict__ ph) {
  __shared__ float buf[8];
  int row = blockIdx.x, t = threadIdx.x;
  int b = row >> 10, l = row & 1023;
  float h0 = (float)h[(size_t)row * 512 + t], h1 = (float)h[(size_t)row * 512 + t + 256];
  for (int j = 0; j < 8; j++) {
    float p = h0 * pw[j * 512 + t] + h1 * pw[j * 512 + t + 256];
    float s = block_reduce_sum(p, buf);
    if (t == 0) ph[((size_t)b * 8 + j) * 1024 + l] = s + pb[j];
  }
}

// ---------------------------------------------------------------- gate + x*g
__global__ void __launch_bounds__(256)
gate_kernel(const _Float16* __restrict__ tb, const float* __restrict__ w2,
            const float* __restrict__ b2, const float* __restrict__ x,
            float* __restrict__ gate, _Float16* __restrict__ xg) {
  __shared__ float buf[8];
  int row = blockIdx.x, t = threadIdx.x;
  float t0 = (float)tb[(size_t)row * 512 + t], t1 = (float)tb[(size_t)row * 512 + t + 256];
  float s = block_reduce_sum(t0 * w2[t] + t1 * w2[t + 256], buf);
  float gv = 1.f / (1.f + __expf(-(s + b2[0])));
  if (t == 0) gate[row] = gv;
  xg[(size_t)row * 512 + t]       = (_Float16)(x[(size_t)row * 512 + t] * gv);
  xg[(size_t)row * 512 + t + 256] = (_Float16)(x[(size_t)row * 512 + t + 256] * gv);
}

// ------------------------------------------------ tiled WMMA GEMM  C=act(A W^T + b)
// A: [M,K] f16 row-major; W: [N,K] f16 row-major; block tile 128x64, BK=32.
// TDM double-buffered staging: wave0 issues tensor_load_to_lds for tile k+1 while
// all 8 waves run WMMA on tile k; one barrier per K-step.
#define G_LD 40  // padded LDS row stride in halves (80B: conflict-free, 16B aligned)

template <int ACT, typename OT>
__global__ void __launch_bounds__(256)
gemm_wmma_kernel(const _Float16* __restrict__ A, const _Float16* __restrict__ W,
                 const float* __restrict__ bias, OT* __restrict__ C,
                 int M, int N, int K) {
  __shared__ __align__(128) _Float16 As[2][128 * G_LD];
  __shared__ __align__(128) _Float16 Ws[2][64 * G_LD];

  const int tid = threadIdx.x;
  const int lane = tid & 31, wave = tid >> 5;
  const int g = lane >> 4, ln = lane & 15;
  const int wm = wave & 3, wn = wave >> 2;           // 4x2 wave grid, 32x32 each
  const int m0 = blockIdx.y * 128;
  const int n0 = blockIdx.x * 64;
  const bool issuer = (wave == 0);

  v8f acc[2][2];
  #pragma unroll
  for (int i = 0; i < 2; i++)
    #pragma unroll
    for (int j = 0; j < 2; j++) acc[i][j] = zero8();

#if USE_TDM
  auto issue = [&](int kb, int bufi) {
    tdm_load_2d_f16(A + (size_t)m0 * K + kb, (unsigned)(uintptr_t)&As[bufi][0],
                    (unsigned)(K - kb), (unsigned)(M - m0), 32u, 128u, (unsigned)K, 3u);
    tdm_load_2d_f16(W + (size_t)n0 * K + kb, (unsigned)(uintptr_t)&Ws[bufi][0],
                    (unsigned)(K - kb), (unsigned)(N - n0), 32u, 64u, (unsigned)K, 3u);
  };
  if (issuer) issue(0, 0);
  const int nk = K >> 5;
  for (int i = 0; i < nk; i++) {
    if (issuer) __builtin_amdgcn_s_wait_tensorcnt(0);
    __syncthreads();  // tile i visible; prior reads of buf[(i+1)&1] done
    if (issuer && (i + 1 < nk)) issue((i + 1) << 5, (i + 1) & 1);
    const _Float16* Ab = &As[i & 1][0];
    const _Float16* Wb = &Ws[i & 1][0];
    v16h af[2], bf[2];
    #pragma unroll
    for (int ii = 0; ii < 2; ii++) af[ii] = frag_a(Ab, G_LD, wm * 32 + ii * 16 + ln, g);
    #pragma unroll
    for (int jj = 0; jj < 2; jj++) bf[jj] = frag_b(Wb, G_LD, wn * 32 + jj * 16 + ln, g);
    #pragma unroll
    for (int ii = 0; ii < 2; ii++)
      #pragma unroll
      for (int jj = 0; jj < 2; jj++)
        acc[ii][jj] = wmma_f16(af[ii], bf[jj], acc[ii][jj]);
  }
#else
  const int arow = tid >> 1, acol = (tid & 1) * 16;
  const int wrow = tid >> 2, wcol = (tid & 3) * 8;
  for (int kb = 0; kb < K; kb += 32) {
    {
      const _Float16* src = A + (size_t)(m0 + arow) * K + kb + acol;
      v8h a0 = *(const v8h*)src;
      v8h a1 = *(const v8h*)(src + 8);
      *(v8h*)(&As[0][0] + arow * G_LD + acol) = a0;
      *(v8h*)(&As[0][0] + arow * G_LD + acol + 8) = a1;
    }
    {
      v8h wv;
      if (n0 + wrow < N) {
        wv = *(const v8h*)(W + (size_t)(n0 + wrow) * K + kb + wcol);
      } else {
        #pragma unroll
        for (int e = 0; e < 8; e++) wv[e] = (_Float16)0.f;
      }
      *(v8h*)(&Ws[0][0] + wrow * G_LD + wcol) = wv;
    }
    __syncthreads();
    v16h af[2], bf[2];
    #pragma unroll
    for (int ii = 0; ii < 2; ii++) af[ii] = frag_a(&As[0][0], G_LD, wm * 32 + ii * 16 + ln, g);
    #pragma unroll
    for (int jj = 0; jj < 2; jj++) bf[jj] = frag_b(&Ws[0][0], G_LD, wn * 32 + jj * 16 + ln, g);
    #pragma unroll
    for (int ii = 0; ii < 2; ii++)
      #pragma unroll
      for (int jj = 0; jj < 2; jj++)
        acc[ii][jj] = wmma_f16(af[ii], bf[jj], acc[ii][jj]);
    __syncthreads();
  }
#endif

  #pragma unroll
  for (int i = 0; i < 2; i++)
    #pragma unroll
    for (int j = 0; j < 2; j++) {
      int col = n0 + wn * 32 + j * 16 + ln;
      if (col < N) {
        float bv = bias ? bias[col] : 0.f;
        #pragma unroll
        for (int r = 0; r < 8; r++) {
          int row = m0 + wm * 32 + i * 16 + r + 8 * g;
          float v = acc[i][j][r] + bv;
          if (ACT == 1) v = 0.5f * v * (1.f + erff(v * 0.70710678118f));  // exact gelu
          else if (ACT == 2) v = tanhf(v);
          C[(size_t)row * N + col] = (OT)v;
        }
      }
    }
}

// -------------------------------------------- flash attention with phase bias
// grid: (L/64, B*H); block: 128 (4 waves, 16 queries each). DH=64, key blocks of 32.
// K block staged by TDM (row stride 72 via pad feature); V^T staged manually.
__global__ void __launch_bounds__(128)
attn_kernel(const _Float16* __restrict__ qkv, const float* __restrict__ phase,
            _Float16* __restrict__ ao) {
  __shared__ __align__(128) _Float16 Kl[32 * 72];   // K block   [key][dh], stride 72
  __shared__ __align__(128) _Float16 Vt[64 * 40];   // V^T block [dh][key], stride 40
  __shared__ __align__(128) _Float16 Pl[4 * 16 * 40];  // P per wave [q][key], stride 40

  const int tid = threadIdx.x, lane = tid & 31, wave = tid >> 5;
  const int g = lane >> 4, ln = lane & 15;
  const int b = blockIdx.y >> 3, hh = blockIdx.y & 7;
  const int q0 = blockIdx.x * 64 + wave * 16;
  const size_t rowbase = (size_t)b * 1024;
  const int phb = (b * 8 + hh) * 1024;
  _Float16* Pw = Pl + wave * 16 * 40;
  const float SCALE = 0.125f;  // 1/sqrt(64)

  // Q A-fragments, loaded once straight from global (16B-aligned b128 pairs)
  v16h aq[2];
  #pragma unroll
  for (int f = 0; f < 2; f++)
    aq[f] = frag_a(qkv + (rowbase + q0) * 1536 + hh * 64 + f * 32, 1536, ln, g);

  float phq[8];
  #pragma unroll
  for (int r = 0; r < 8; r++) phq[r] = phase[phb + q0 + r + 8 * g];

  v8f o[4];
  #pragma unroll
  for (int j = 0; j < 4; j++) o[j] = zero8();
  float m_row[8], l_row[8];
  #pragma unroll
  for (int r = 0; r < 8; r++) { m_row[r] = -1e30f; l_row[r] = 0.f; }

  const int srow = tid >> 2, schunk = tid & 3;

  for (int kb = 0; kb < 1024; kb += 32) {
    __syncthreads();  // previous iteration's LDS reads complete
#if USE_TDM
    if (wave == 0) {  // stage K block [32 keys][64 dh] via TDM, stride 72 halves
      tdm_load_2d_f16(qkv + (rowbase + kb) * 1536 + 512 + hh * 64,
                      (unsigned)(uintptr_t)&Kl[0],
                      (unsigned)(1536 - (512 + hh * 64)),
                      (unsigned)(4096 - (unsigned)(rowbase + kb)),
                      64u, 32u, 1536u, 4u);
    }
#else
    {  // stage K block [32][64]
      const _Float16* src = qkv + (rowbase + kb + srow) * 1536 + 512 + hh * 64 + schunk * 16;
      *(v8h*)(Kl + srow * 72 + schunk * 16) = *(const v8h*)src;
      *(v8h*)(Kl + srow * 72 + schunk * 16 + 8) = *(const v8h*)(src + 8);
    }
#endif
    {  // stage V^T block [64][32]
      const _Float16* src = qkv + (rowbase + kb + srow) * 1536 + 1024 + hh * 64 + schunk * 16;
      HV u;
      u.v8[0] = *(const v8h*)src;
      u.v8[1] = *(const v8h*)(src + 8);
      #pragma unroll
      for (int e = 0; e < 16; e++) Vt[(schunk * 16 + e) * 40 + srow] = u.v16[e];
    }
#if USE_TDM
    if (wave == 0) __builtin_amdgcn_s_wait_tensorcnt(0);
#endif
    __syncthreads();

    // S = Q K^T  (two 16x16 key tiles, K-dim = 64 over two WMMA steps)
    v8f s0 = zero8(), s1 = zero8();
    #pragma unroll
    for (int ks = 0; ks < 2; ks++) {
      v16h b0 = frag_b(Kl + ks * 32, 72, ln, g);
      v16h b1 = frag_b(Kl + ks * 32, 72, 16 + ln, g);
      s0 = wmma_f16(aq[ks], b0, s0);
      s1 = wmma_f16(aq[ks], b1, s1);
    }

    float pk0 = phase[phb + kb + ln];
    float pk1 = phase[phb + kb + 16 + ln];
    float alpha_r[8];
    #pragma unroll
    for (int r = 0; r < 8; r++) {
      float pq = phq[r];
      float d0 = pq - pk0, d1 = pq - pk1;
      float s0v = s0[r] * SCALE - d0 * d0;
      float s1v = s1[r] * SCALE - d1 * d1;
      float cur = fmaxf(s0v, s1v);
      #pragma unroll
      for (int off = 1; off < 16; off <<= 1) cur = fmaxf(cur, __shfl_xor(cur, off, 32));
      float mnew = fmaxf(m_row[r], cur);
      float p0 = __expf(s0v - mnew);
      float p1 = __expf(s1v - mnew);
      float rs = p0 + p1;
      #pragma unroll
      for (int off = 1; off < 16; off <<= 1) rs += __shfl_xor(rs, off, 32);
      float alpha = __expf(m_row[r] - mnew);
      l_row[r] = l_row[r] * alpha + rs;
      m_row[r] = mnew;
      alpha_r[r] = alpha;
      Pw[(r + 8 * g) * 40 + ln] = (_Float16)p0;
      Pw[(r + 8 * g) * 40 + 16 + ln] = (_Float16)p1;
    }
    #pragma unroll
    for (int j = 0; j < 4; j++) {
      v8f t = o[j];
      #pragma unroll
      for (int r = 0; r < 8; r++) t[r] *= alpha_r[r];
      o[j] = t;
    }
    __syncthreads();  // P visible to all lanes of the wave (uniform across block)

    // O += P V
    v16h pf = frag_a(Pw, 40, ln, g);
    #pragma unroll
    for (int j = 0; j < 4; j++) {
      v16h bv = frag_b(Vt, 40, j * 16 + ln, g);
      o[j] = wmma_f16(pf, bv, o[j]);
    }
  }

  #pragma unroll
  for (int r = 0; r < 8; r++) {
    float inv = 1.f / l_row[r];
    int q = q0 + r + 8 * g;
    #pragma unroll
    for (int j = 0; j < 4; j++)
      ao[(rowbase + q) * 512 + hh * 64 + j * 16 + ln] = (_Float16)(o[j][r] * inv);
  }
}

// -----------------------------------------------------------------------------
extern "C" void kernel_launch(void* const* d_in, const int* in_sizes, int n_in,
                              void* d_out, int out_size, void* d_ws, size_t ws_size,
                              hipStream_t stream) {
  (void)in_sizes; (void)n_in; (void)out_size; (void)ws_size;
  const int*   ids    = (const int*)  d_in[0];
  const float* embedw = (const float*)d_in[1];
  const float* pos    = (const float*)d_in[2];
  const float* rg_w   = (const float*)d_in[3];
  const float* rg_b   = (const float*)d_in[4];
  const float* qkv_w  = (const float*)d_in[5];
  const float* qkv_b  = (const float*)d_in[6];
  const float* out_w  = (const float*)d_in[7];
  const float* out_b  = (const float*)d_in[8];
  const float* ph_w   = (const float*)d_in[9];
  const float* ph_b   = (const float*)d_in[10];
  const float* ff1_w  = (const float*)d_in[11];
  const float* ff1_b  = (const float*)d_in[12];
  const float* ff2_w  = (const float*)d_in[13];
  const float* ff2_b  = (const float*)d_in[14];
  const float* n1_s   = (const float*)d_in[15];
  const float* n1_b   = (const float*)d_in[16];
  const float* n2_s   = (const float*)d_in[17];
  const float* n2_b   = (const float*)d_in[18];
  const float* bnd_w1 = (const float*)d_in[19];
  const float* bnd_b1 = (const float*)d_in[20];
  const float* bnd_w2 = (const float*)d_in[21];
  const float* bnd_b2 = (const float*)d_in[22];
  const float* head_w = (const float*)d_in[23];

  char* ws = (char*)d_ws;
  size_t off = 0;
  auto take = [&](size_t bytes) -> char* {
    char* p = ws + off;
    off = (off + bytes + 255) & ~(size_t)255;
    return p;
  };
  float*    x     = (float*)   take((size_t)4096 * 512 * 4);
  _Float16* h16   = (_Float16*)take((size_t)4096 * 512 * 2);
  _Float16* qkvb  = (_Float16*)take((size_t)4096 * 1536 * 2);
  float*    phbuf = (float*)   take((size_t)4 * 8 * 1024 * 4);
  _Float16* aob   = (_Float16*)take((size_t)4096 * 512 * 2);
  float*    tmp   = (float*)   take((size_t)4096 * 512 * 4);
  _Float16* ffh   = (_Float16*)take((size_t)4096 * 2048 * 2);
  _Float16* tb    = (_Float16*)take((size_t)4096 * 512 * 2);
  _Float16* xg    = (_Float16*)take((size_t)4096 * 512 * 2);
  _Float16* wqkv  = (_Float16*)take((size_t)6 * 1536 * 512 * 2);
  _Float16* wout  = (_Float16*)take((size_t)6 * 512 * 512 * 2);
  _Float16* wff1  = (_Float16*)take((size_t)6 * 2048 * 512 * 2);
  _Float16* wff2  = (_Float16*)take((size_t)6 * 512 * 2048 * 2);
  _Float16* wbnd  = (_Float16*)take((size_t)512 * 512 * 2);
  _Float16* whead = (_Float16*)take((size_t)30000 * 512 * 2);

  float* logits = (float*)d_out;
  float* gate   = logits + (size_t)4096 * 30000;
  float* rgout  = gate + 4096;

  auto conv = [&](const float* s, _Float16* d, long n) {
    f32_to_f16_kernel<<<2048, 256, 0, stream>>>(s, d, (int)n);
  };
  conv(qkv_w,  wqkv,  6L * 1536 * 512);
  conv(out_w,  wout,  6L * 512 * 512);
  conv(ff1_w,  wff1,  6L * 2048 * 512);
  conv(ff2_w,  wff2,  6L * 512 * 2048);
  conv(bnd_w1, wbnd,  512L * 512);
  conv(head_w, whead, 30000L * 512);

  embed_kernel<<<(4096 * 512 + 255) / 256, 256, 0, stream>>>(ids, embedw, pos, x);

  for (int l = 0; l < 6; l++) {
    rg_mod_ln_kernel<<<4096, 256, 0, stream>>>(
        x, rg_w + l * 4 * 512, rg_b + l * 4, n1_s + l * 512, n1_b + l * 512,
        h16, (l == 5) ? rgout : nullptr);
    gemm_wmma_kernel<0, _Float16><<<dim3(1536 / 64, 32), 256, 0, stream>>>(
        h16, wqkv + (size_t)l * 1536 * 512, qkv_b + l * 1536, qkvb, 4096, 1536, 512);
    phase_kernel<<<4096, 256, 0, stream>>>(h16, ph_w + l * 8 * 512, ph_b + l * 8, phbuf);
    attn_kernel<<<dim3(16, 32), 128, 0, stream>>>(qkvb, phbuf, aob);
    gemm_wmma_kernel<0, float><<<dim3(512 / 64, 32), 256, 0, stream>>>(
        aob, wout + (size_t)l * 512 * 512, out_b + l * 512, tmp, 4096, 512, 512);
    add_ln_kernel<<<4096, 256, 0, stream>>>(x, tmp, n1_s + l * 512, n1_b + l * 512, h16);
    gemm_wmma_kernel<1, _Float16><<<dim3(2048 / 64, 32), 256, 0, stream>>>(
        h16, wff1 + (size_t)l * 2048 * 512, ff1_b + l * 2048, ffh, 4096, 2048, 512);
    gemm_wmma_kernel<0, float><<<dim3(512 / 64, 32), 256, 0, stream>>>(
        ffh, wff2 + (size_t)l * 512 * 2048, ff2_b + l * 512, tmp, 4096, 512, 2048);
    add_ln_kernel<<<4096, 256, 0, stream>>>(x, tmp, n2_s + l * 512, n2_b + l * 512, h16);
  }

  gemm_wmma_kernel<2, _Float16><<<dim3(512 / 64, 32), 256, 0, stream>>>(
      h16, wbnd, bnd_b1, tb, 4096, 512, 512);
  gate_kernel<<<4096, 256, 0, stream>>>(tb, bnd_w2, bnd_b2, x, gate, xg);
  gemm_wmma_kernel<0, float><<<dim3((30000 + 63) / 64, 32), 256, 0, stream>>>(
      xg, whead, nullptr, logits, 4096, 30000, 512);
}
